// CrossViewInteraction_59622736003366
// MI455X (gfx1250) — matrix operand order repaired and checked by reading, not compiled, not run on hardware
//
#include <hip/hip_runtime.h>
#include <hip/hip_bf16.h>

#define VV    4
#define NN    1536
#define DDIM  256
#define TOPKN 32
#define NH    4
#define HDIM  64
#define MM    (VV*NN)

typedef __attribute__((ext_vector_type(16))) __bf16 v16bf;
typedef __attribute__((ext_vector_type(8)))  float  v8f;

__device__ __forceinline__ uint32_t pk_bf16(float lo, float hi) {
  union { __bf16 h[2]; uint32_t u; } c;
  c.h[0] = (__bf16)lo; c.h[1] = (__bf16)hi;
  return c.u;                       // lowers to v_cvt_pk_bf16_f32
}

// ---------------------------------------------------------------------------
// GEMM: Out[M x Nc] = A[M x K] @ W[Nc x K]^T (+bias / epilogue), bf16 WMMA.
// 256-thread block = 8 waves; macro-tile 64(M) x 64(N).
// Each wave owns one A fragment and TWO B fragments -> 2 WMMA per k-step
// (16 in the unrolled kernel), sharing the A gather and LDS stage.
// A/B staged in LDS as packed bf16 pairs (u32); fragment gathers are 8 b32
// reads each (CDNA5 layouts keep K-pairs adjacent).
// mode 0: Out = t + bias
// mode 1: Out = (t + bias)*alphas[v] + (1-alphas[v])*Hsrc   (aligned blend)
// mode 2: Out = t                                            (per-view W)
// ---------------------------------------------------------------------------
__global__ __launch_bounds__(256)
void gemm_xwt_wmma(const float* __restrict__ A, const float* __restrict__ W,
                   const float* __restrict__ bias, float* __restrict__ Out,
                   int M, int K, int Nc, int perViewW, int mode,
                   const float* __restrict__ Hsrc, const float* __restrict__ alphas)
{
  __shared__ uint32_t lA[64][17];   // 64 rows x 16 K-pairs (+1 pad)
  __shared__ uint32_t lB[64][17];   // 64 rows x 16 K-pairs (+1 pad)
  const int t       = threadIdx.x;
  const int lane    = t & 31;
  const int w       = t >> 5;          // wave 0..7
  const int wm      = w & 3;           // 0..3  (M sub-tile)
  const int wn2     = w >> 2;          // 0..1  (pair of N sub-tiles)
  const int rowBase = blockIdx.y * 64;
  const int colBase = blockIdx.x * 64;
  const int view    = rowBase / NN;    // 1536 % 64 == 0 -> uniform per block
  const float* Wp   = perViewW ? (W + (size_t)view * K * Nc) : W;

  v8f acc0 = {}, acc1 = {};
  const int m     = lane & 15;         // M row (A) / N col (B) within tile
  const int khalf = lane >> 4;

  #pragma unroll
  for (int kt = 0; kt < 256; kt += 32) {
    // cooperative fill: A 64x32 and B 64x32, 1024 bf16-pairs each, 4/thread
    #pragma unroll
    for (int i = 0; i < 4; ++i) {
      int f = t + 256 * i;             // 0..1023 pair index
      int r = f >> 4, pc = f & 15;
      const float2 xa = *reinterpret_cast<const float2*>(
          A + (size_t)(rowBase + r) * K + kt + 2 * pc);
      lA[r][pc] = pk_bf16(xa.x, xa.y);
      const float2 xb = *reinterpret_cast<const float2*>(
          Wp + (size_t)(colBase + r) * K + kt + 2 * pc);
      lB[r][pc] = pk_bf16(xb.x, xb.y);
    }
    __syncthreads();

    // Fragment gather in contiguous k order shared by A and B (pairs stay adjacent -> 8 b32 reads each).
    // A 16x32: K 0..15 lanes0-15, 16..31 lanes16-31
    // B 32x16: pair p -> rows 2p,2p+1; K 0..15 lanes0-15, 16..31 lanes16-31
    union { uint32_t u[8]; v16bf v; } af, bf0, bf1;
    #pragma unroll
    for (int p = 0; p < 8; ++p) {
      int kpA = khalf * 8 + p;
      af.u[p]  = lA[wm * 16 + m][kpA];
      int kpB = khalf * 8 + p;
      bf0.u[p] = lB[wn2 * 32 + m][kpB];
      bf1.u[p] = lB[wn2 * 32 + 16 + m][kpB];
    }
    acc0 = __builtin_amdgcn_wmma_f32_16x16x32_bf16(false, af.v, false, bf0.v,
                                                   (short)0, acc0, false, false);
    acc1 = __builtin_amdgcn_wmma_f32_16x16x32_bf16(false, af.v, false, bf1.v,
                                                   (short)0, acc1, false, false);
    __syncthreads();
  }

  // C/D layout: VGPR r -> M=r (lanes 0-15) / M=8+r (lanes 16-31), N = lane&15
  const int nloc = lane & 15;
  #pragma unroll
  for (int r = 0; r < 8; ++r) {
    int mloc = r + 8 * khalf;
    int row  = rowBase + wm * 16 + mloc;
    #pragma unroll
    for (int nb = 0; nb < 2; ++nb) {
      int col = colBase + wn2 * 32 + nb * 16 + nloc;
      float val = nb ? acc1[r] : acc0[r];
      if (mode == 0) {
        val += bias[col];
      } else if (mode == 1) {
        val += bias[col];
        float a = alphas[row / NN];
        val = val * a + (1.0f - a) * Hsrc[(size_t)row * Nc + col];
      }
      Out[(size_t)row * Nc + col] = val;
    }
  }
}

// ---------------------------------------------------------------------------
// Attention over the VIEW axis (length 4) per (n, head). One wave per (n,h),
// each lane owns 2 of the 64 head dims. Cross-lane dot via shfl_xor (wave32).
// ---------------------------------------------------------------------------
__global__ __launch_bounds__(32)
void attn4_views(const float* __restrict__ qkv, float* __restrict__ o)
{
  const int wid  = blockIdx.x;
  const int n    = wid / NH;
  const int h    = wid % NH;
  const int lane = threadIdx.x;
  const int d0   = 2 * lane;

  float q[VV][2], k[VV][2], vv[VV][2];
  #pragma unroll
  for (int i = 0; i < VV; ++i) {
    const float* base = qkv + (size_t)(i * NN + n) * (3 * DDIM) + h * HDIM;
    q [i][0] = base[d0];            q [i][1] = base[d0 + 1];
    k [i][0] = base[DDIM + d0];     k [i][1] = base[DDIM + d0 + 1];
    vv[i][0] = base[2*DDIM + d0];   vv[i][1] = base[2*DDIM + d0 + 1];
  }
  float s[VV][VV];
  #pragma unroll
  for (int i = 0; i < VV; ++i)
    #pragma unroll
    for (int j = 0; j < VV; ++j) {
      float part = q[i][0] * k[j][0] + q[i][1] * k[j][1];
      #pragma unroll
      for (int off = 16; off > 0; off >>= 1) part += __shfl_xor(part, off, 32);
      s[i][j] = part * 0.125f;   // 1/sqrt(HD=64)
    }
  #pragma unroll
  for (int i = 0; i < VV; ++i) {
    float mx = s[i][0];
    #pragma unroll
    for (int j = 1; j < VV; ++j) mx = fmaxf(mx, s[i][j]);
    float sum = 0.0f;
    #pragma unroll
    for (int j = 0; j < VV; ++j) { s[i][j] = expf(s[i][j] - mx); sum += s[i][j]; }
    float inv = 1.0f / sum;
    float o0 = 0.0f, o1 = 0.0f;
    #pragma unroll
    for (int j = 0; j < VV; ++j) { o0 += s[i][j] * vv[j][0]; o1 += s[i][j] * vv[j][1]; }
    float* ob = o + (size_t)(i * NN + n) * DDIM + h * HDIM;
    ob[d0]     = o0 * inv;
    ob[d0 + 1] = o1 * inv;
  }
}

// ---------------------------------------------------------------------------
// Column mean of Vn per view: exact diagonal (p==q) softmax contribution,
// since fully-masked rows give uniform 1/N weights.
// ---------------------------------------------------------------------------
__global__ __launch_bounds__(256)
void vmean_kernel(const float* __restrict__ Vn, float* __restrict__ Vmean)
{
  const int v = blockIdx.x, d = threadIdx.x;
  float s = 0.0f;
  for (int mI = 0; mI < NN; ++mI) s += Vn[(size_t)(v * NN + mI) * DDIM + d];
  Vmean[v * DDIM + d] = s * (1.0f / NN);
}

// ---------------------------------------------------------------------------
// Top-32 indices of C[p,q,n,:] (p != q only). One wave per row; row cached in
// LDS; 32 rounds of wave argmax (tie -> lowest index, matching jax top_k).
// ---------------------------------------------------------------------------
__global__ __launch_bounds__(32)
void topk_kernel(const float* __restrict__ C, int* __restrict__ idxOut)
{
  __shared__ float sv[NN];
  const int rid = blockIdx.x;
  const int n   = rid % NN;
  const int pq  = rid / NN;
  const int p   = pq / (VV - 1);
  const int qq  = pq % (VV - 1);
  const int q   = qq + (qq >= p ? 1 : 0);
  const int lane = threadIdx.x;

  const float* row = C + ((size_t)((p * VV + q) * NN) + n) * NN;
  #pragma unroll
  for (int j = 0; j < NN / 32; ++j) sv[lane + 32 * j] = row[lane + 32 * j];
  __syncthreads();

  int* ob = idxOut + ((size_t)((p * VV + q) * NN) + n) * TOPKN;
  for (int it = 0; it < TOPKN; ++it) {
    float bv = -3.4e38f; int bi = 0x7FFFFFFF;
    for (int j = 0; j < NN / 32; ++j) {
      float x = sv[lane + 32 * j];
      int  gi = lane + 32 * j;
      if (x > bv) { bv = x; bi = gi; }
    }
    #pragma unroll
    for (int off = 16; off > 0; off >>= 1) {
      float ov = __shfl_xor(bv, off, 32);
      int   oi = __shfl_xor(bi, off, 32);
      if (ov > bv || (ov == bv && oi < bi)) { bv = ov; bi = oi; }
    }
    if (lane == 0) ob[it] = bi;
    if (lane == (bi & 31)) sv[bi] = -3.4e38f;   // single-lane LDS clear
  }
}

// ---------------------------------------------------------------------------
// Sparse neighbor attention + final gating. One 256-thread block per (p,n).
// 96 gathered score rows (3 views x 32 neighbors), exact sparse softmax,
// diagonal added via Vmean. Epilogue fuses relu/sigmoid gates.
// ---------------------------------------------------------------------------
__global__ __launch_bounds__(256)
void sparse_attn_final(const float* __restrict__ Qn, const float* __restrict__ Kn,
                       const float* __restrict__ Vn, const float* __restrict__ Vmean,
                       const int* __restrict__ idx,  const float* __restrict__ aligned,
                       const float* __restrict__ Hin,
                       const float* __restrict__ alpha_align,
                       const float* __restrict__ beta, float* __restrict__ out)
{
  __shared__ float sQ[DDIM];
  __shared__ float sw[96];
  __shared__ int   si[96];
  const int bid = blockIdx.x;
  const int p   = bid / NN;
  const int n   = bid % NN;
  const int t   = threadIdx.x;
  const int lane = t & 31, wv = t >> 5;

  sQ[t] = Qn[(size_t)(p * NN + n) * DDIM + t];
  __syncthreads();

  // 96 score rows, 12 per wave: dot(Q, gathered K row) / sqrt(D)
  for (int pr = wv * 12; pr < wv * 12 + 12; ++pr) {
    int qq = pr >> 5;
    int q  = qq + (qq >= p ? 1 : 0);
    int j  = pr & 31;
    int mI = idx[(((size_t)(p * VV + q)) * NN + n) * TOPKN + j];
    const float* krow = Kn + (size_t)(q * NN + mI) * DDIM;
    float part = 0.0f;
    #pragma unroll
    for (int e = 0; e < 8; ++e) { int d = e * 32 + lane; part += sQ[d] * krow[d]; }
    #pragma unroll
    for (int off = 16; off > 0; off >>= 1) part += __shfl_xor(part, off, 32);
    if (lane == 0) { sw[pr] = part * (1.0f / 16.0f); si[pr] = mI; }
  }
  __syncthreads();

  if (t < VV - 1) {        // per-q softmax over 32 sparse scores
    float mx = -3.4e38f;
    for (int j = 0; j < 32; ++j) mx = fmaxf(mx, sw[t * 32 + j]);
    float s = 0.0f;
    for (int j = 0; j < 32; ++j) { float e = expf(sw[t * 32 + j] - mx); sw[t * 32 + j] = e; s += e; }
    float inv = 1.0f / s;
    for (int j = 0; j < 32; ++j) sw[t * 32 + j] *= inv;
  }
  __syncthreads();

  float acc = Vmean[p * DDIM + t];          // exact diagonal contribution
  for (int pr = 0; pr < 96; ++pr) {
    int qq = pr >> 5;
    int q  = qq + (qq >= p ? 1 : 0);
    acc += sw[pr] * Vn[(size_t)(q * NN + si[pr]) * DDIM + t];
  }

  const float aa = 1.0f / (1.0f + expf(-alpha_align[0]));
  const float bb = 1.0f / (1.0f + expf(-beta[0]));
  const size_t off = (size_t)(p * NN + n) * DDIM + t;
  float f = aa * aligned[off] + (1.0f - aa) * acc;
  f = fmaxf(f, 0.0f);
  out[off] = bb * Hin[off] + (1.0f - bb) * f;
}

// ---------------------------------------------------------------------------
extern "C" void kernel_launch(void* const* d_in, const int* in_sizes, int n_in,
                              void* d_out, int out_size, void* d_ws, size_t ws_size,
                              hipStream_t stream)
{
  const float* H    = (const float*)d_in[0];
  const float* C    = (const float*)d_in[1];
  const float* WQ   = (const float*)d_in[2];
  const float* WK   = (const float*)d_in[3];
  const float* WV   = (const float*)d_in[4];
  const float* ipw  = (const float*)d_in[5];
  const float* ipb  = (const float*)d_in[6];
  const float* ow   = (const float*)d_in[7];
  const float* ob   = (const float*)d_in[8];
  const float* alphas  = (const float*)d_in[9];
  const float* a_align = (const float*)d_in[10];
  const float* betap   = (const float*)d_in[11];
  float* out = (float*)d_out;

  // workspace layout (floats); qkv region is reused for Qn/Kn/Vn once dead
  float* ws      = (float*)d_ws;
  float* qkv     = ws;                                  // MM*768
  float* qn      = ws;                                  // MM*256 (reuse)
  float* kn      = ws + (size_t)MM * DDIM;              // MM*256 (reuse)
  float* vn      = ws + (size_t)2 * MM * DDIM;          // MM*256 (reuse)
  float* obuf    = ws + (size_t)MM * 3 * DDIM;          // MM*256
  float* alignedb= obuf + (size_t)MM * DDIM;            // MM*256
  float* vmean   = alignedb + (size_t)MM * DDIM;        // VV*256
  int*   idxb    = (int*)(vmean + VV * DDIM);           // VV*VV*NN*32

  // 1. qkv = H @ in_proj_w^T + b
  gemm_xwt_wmma<<<dim3(3 * DDIM / 64, MM / 64), 256, 0, stream>>>(
      H, ipw, ipb, qkv, MM, DDIM, 3 * DDIM, 0, 0, nullptr, nullptr);

  // 2. 4-wide attention over views
  attn4_views<<<NN * NH, 32, 0, stream>>>(qkv, obuf);

  // 3. aligned = (o @ out_w^T + b)*a + (1-a)*H
  gemm_xwt_wmma<<<dim3(DDIM / 64, MM / 64), 256, 0, stream>>>(
      obuf, ow, ob, alignedb, MM, DDIM, DDIM, 0, 1, H, alphas);

  // 4. Qn/Kn/Vn = aligned @ W{Q,K,V}[v]^T   (per-view weights)
  gemm_xwt_wmma<<<dim3(DDIM / 64, MM / 64), 256, 0, stream>>>(
      alignedb, WQ, nullptr, qn, MM, DDIM, DDIM, 1, 2, nullptr, nullptr);
  gemm_xwt_wmma<<<dim3(DDIM / 64, MM / 64), 256, 0, stream>>>(
      alignedb, WK, nullptr, kn, MM, DDIM, DDIM, 1, 2, nullptr, nullptr);
  gemm_xwt_wmma<<<dim3(DDIM / 64, MM / 64), 256, 0, stream>>>(
      alignedb, WV, nullptr, vn, MM, DDIM, DDIM, 1, 2, nullptr, nullptr);

  // 5. diagonal softmax contribution = column mean of Vn
  vmean_kernel<<<VV, 256, 0, stream>>>(vn, vmean);

  // 6. top-32 indices of C rows (p != q)
  topk_kernel<<<VV * (VV - 1) * NN, 32, 0, stream>>>(C, idxb);

  // 7. sparse neighbor attention + gated output
  sparse_attn_final<<<VV * NN, 256, 0, stream>>>(
      qn, kn, vn, vmean, idxb, alignedb, H, a_align, betap, out);
}
